// CustomBlock_56264071577748
// MI455X (gfx1250) — matrix operand order repaired and checked by reading, not compiled, not run on hardware
//
#include <hip/hip_runtime.h>

#define NB 8
#define CD 2048
#define BD 2048

typedef __attribute__((ext_vector_type(16))) __bf16 v16bf;
typedef __attribute__((ext_vector_type(8)))  __bf16 v8bf;
typedef __attribute__((ext_vector_type(4)))  __bf16 v4bf;
typedef __attribute__((ext_vector_type(8)))  float  v8f;
typedef unsigned int  u32x4 __attribute__((ext_vector_type(4)));
typedef int           i32x4 __attribute__((ext_vector_type(4)));
typedef int           i32x8 __attribute__((ext_vector_type(8)));

// ---------------------------------------------------------------- WMMA core
static __device__ __forceinline__ v8f wmma_bf16(v16bf a, v16bf b, v8f c) {
    return __builtin_amdgcn_wmma_f32_16x16x32_bf16(
        /*neg_a=*/false, a, /*neg_b=*/false, b,
        /*c_mod=*/(short)0, c, /*reuse_a=*/false, /*reuse_b=*/false);
}

// A fragment from a row-major bf16 matrix (global), tile (row0,k0), leading dim ld.
// Lane m=lane&15 holds row m; kh=lane>>4: elems 0..7 <-> K=8*kh+0..7, 8..15 <-> K=16+8*kh+0..7
static __device__ __forceinline__ v16bf load_a(const __bf16* __restrict__ p, int ld,
                                               int row0, int k0, int lane) {
    int m = lane & 15, kh = lane >> 4;
    const __bf16* base = p + (size_t)(row0 + m) * ld + (k0 + 8 * kh);
    v8bf lo = *reinterpret_cast<const v8bf*>(base);
    v8bf hi = *reinterpret_cast<const v8bf*>(base + 16);
    return __builtin_shufflevector(lo, hi, 0,1,2,3,4,5,6,7,8,9,10,11,12,13,14,15);
}

// B fragment B[k][n] sourced from BT row-major (BT[n][k]), leading dim ld.
static __device__ __forceinline__ v16bf load_bt(const __bf16* __restrict__ pT, int ld,
                                                int col0, int k0, int lane) {
    int nn = lane & 15, kh = lane >> 4;
    const __bf16* base = pT + (size_t)(col0 + nn) * ld + (k0 + 8 * kh);
    v8bf lo = *reinterpret_cast<const v8bf*>(base);
    v8bf hi = *reinterpret_cast<const v8bf*>(base + 16);
    return __builtin_shufflevector(lo, hi, 0,1,2,3,4,5,6,7,8,9,10,11,12,13,14,15);
}

// LDS panel fragment loaders. Panel layout: [128 rows][32 K] with 8-element (16B) row pad
// => row stride 40 elements (80B), which is bank-conflict-free across 16 consecutive rows.
#define LDS_STRIDE 40
static __device__ __forceinline__ v16bf lds_frag_a(const __bf16* p, int row0, int lane) {
    int m = lane & 15, kh = lane >> 4;
    const __bf16* base = p + (row0 + m) * LDS_STRIDE + 8 * kh;
    v8bf lo = *reinterpret_cast<const v8bf*>(base);
    v8bf hi = *reinterpret_cast<const v8bf*>(base + 16);
    return __builtin_shufflevector(lo, hi, 0,1,2,3,4,5,6,7,8,9,10,11,12,13,14,15);
}
static __device__ __forceinline__ v16bf lds_frag_b(const __bf16* p, int col0, int lane) {
    int nn = lane & 15, kh = lane >> 4;
    const __bf16* base = p + (col0 + nn) * LDS_STRIDE + 8 * kh;
    v8bf lo = *reinterpret_cast<const v8bf*>(base);
    v8bf hi = *reinterpret_cast<const v8bf*>(base + 16);
    return __builtin_shufflevector(lo, hi, 0,1,2,3,4,5,6,7,8,9,10,11,12,13,14,15);
}

// ---------------------------------------------------------------- Tensor Data Mover
// 2D tile DMA: global (row-major, ld elems, bf16) -> LDS, tile = tile_rows x 32 elems,
// with 16B LDS pad after each 64B row (=> LDS_STRIDE elems per row).
// D# built per CDNA5 ISA ch.8: group0 = {flags, lds_addr, global_addr, type}, group1 = dims.
static __device__ __forceinline__ void tdm_load_tile(const __bf16* gptr, const __bf16* lptr,
                                                     int tile_rows, int ld_elems) {
    unsigned long long ga = (unsigned long long)(size_t)gptr;
    unsigned int laddr = (unsigned int)(size_t)lptr;   // low 32 bits = LDS byte offset

    u32x4 g0;
    g0[0] = 1u;                                             // count=1, user descriptor
    g0[1] = laddr;                                          // lds_addr (bytes)
    g0[2] = (unsigned int)ga;                               // global_addr[31:0]
    g0[3] = (unsigned int)((ga >> 32) & 0x1FFFFFFull)       // global_addr[56:32]
          | (2u << 30);                                     // type = 2 ("image")

    const int tk = 32;                                      // tile_dim0 (elems, 64B row)
    i32x8 g1;
    g1[0] = (1 << 16)                                       // data_size = 2 bytes
          | (1 << 20)                                       // pad_enable
          | (3 << 22)                                       // pad_interval: every 64B
          | (3 << 25);                                      // pad_amount: 4 dwords (16B)
    g1[1] = (tk & 0xFFFF) << 16;                            // tensor_dim0[15:0]
    g1[2] = ((tk >> 16) & 0xFFFF) | ((tile_rows & 0xFFFF) << 16);   // dim0 hi | dim1 lo
    g1[3] = ((tile_rows >> 16) & 0xFFFF) | ((tk & 0xFFFF) << 16);   // dim1 hi | tile_dim0
    g1[4] = (tile_rows & 0xFFFF);                           // tile_dim1 | tile_dim2=0 (2D)
    g1[5] = ld_elems;                                       // tensor_dim0_stride[31:0]
    g1[6] = 0;                                              // stride0 hi | stride1 lo
    g1[7] = 0;

    i32x4 z4 = {0, 0, 0, 0};
#if __has_include(<hip/amd_detail/amd_gfx1250_TDM.h>)
    i32x8 z8 = {0, 0, 0, 0, 0, 0, 0, 0};
    __builtin_amdgcn_tensor_load_to_lds(g0, g1, z4, z4, z8, 0);   // clang-23 / therock arity
#else
    __builtin_amdgcn_tensor_load_to_lds(g0, g1, z4, z4, 0);       // ROCm 7.2 clang-22 arity
#endif
}

// ---------------------------------------------------------------- elementwise cast
__global__ void cast_bf16_kernel(const float* __restrict__ in, __bf16* __restrict__ out,
                                 int count) {
    int i = blockIdx.x * blockDim.x + threadIdx.x;
    if (i < count) out[i] = (__bf16)in[i];
}

// ---------------------------------------------------------------- X -> Xt (bf16, transposed per batch)
__global__ void transpose_cast_kernel(const float* __restrict__ X, __bf16* __restrict__ Xt) {
    __shared__ float tile[32][33];
    int nbat = blockIdx.y;
    const float* Xn = X + (size_t)nbat * CD * BD;
    __bf16* Xtn = Xt + (size_t)nbat * BD * CD;
    const int tiles_x = BD / 32;
    int b0 = (blockIdx.x % tiles_x) * 32;
    int c0 = (blockIdx.x / tiles_x) * 32;
    for (int dy = 0; dy < 32; dy += 8)
        tile[threadIdx.y + dy][threadIdx.x] =
            Xn[(size_t)(c0 + threadIdx.y + dy) * BD + b0 + threadIdx.x];
    __syncthreads();
    for (int dy = 0; dy < 32; dy += 8)
        Xtn[(size_t)(b0 + threadIdx.y + dy) * CD + c0 + threadIdx.x] =
            (__bf16)tile[threadIdx.x][threadIdx.y + dy];
}

// ---------------------------------------------------------------- K = Wk*X + bk, Q = Wq*X + bq
// Direct-stream form; Xt B-fragments shared between the K and Q accumulators.
__global__ void __launch_bounds__(256) gemm_kq_kernel(
    const __bf16* __restrict__ Wkb, const __bf16* __restrict__ Wqb,
    const __bf16* __restrict__ Xt,
    const float* __restrict__ bk, const float* __restrict__ bq,
    __bf16* __restrict__ Kb, __bf16* __restrict__ Qb) {
    int nbat = blockIdx.y;
    const __bf16* Xtn = Xt + (size_t)nbat * BD * CD;
    __bf16* Kn = Kb + (size_t)nbat * CD * BD;
    __bf16* Qn = Qb + (size_t)nbat * CD * BD;

    int lane = threadIdx.x & 31;
    int W = blockIdx.x * (blockDim.x >> 5) + (threadIdx.x >> 5);
    int i0 = (W >> 6) * 32;
    int b0 = (W & 63) * 32;

    v8f aK[2][2], aQ[2][2];
    for (int mi = 0; mi < 2; ++mi)
        for (int nj = 0; nj < 2; ++nj) { aK[mi][nj] = {}; aQ[mi][nj] = {}; }

    #pragma unroll 2
    for (int k0 = 0; k0 < CD; k0 += 32) {
        if (k0 + 64 < CD) {
            __builtin_prefetch(Xtn + (size_t)(b0 + (lane & 15)) * CD + k0 + 64, 0, 1);
            __builtin_prefetch(Wkb + (size_t)(i0 + (lane & 15)) * CD + k0 + 64, 0, 1);
        }
        v16bf ak0 = load_a(Wkb, CD, i0,      k0, lane);
        v16bf ak1 = load_a(Wkb, CD, i0 + 16, k0, lane);
        v16bf aq0 = load_a(Wqb, CD, i0,      k0, lane);
        v16bf aq1 = load_a(Wqb, CD, i0 + 16, k0, lane);
        v16bf bf0 = load_bt(Xtn, CD, b0,      k0, lane);
        v16bf bf1 = load_bt(Xtn, CD, b0 + 16, k0, lane);
        aK[0][0] = wmma_bf16(ak0, bf0, aK[0][0]);
        aK[0][1] = wmma_bf16(ak0, bf1, aK[0][1]);
        aK[1][0] = wmma_bf16(ak1, bf0, aK[1][0]);
        aK[1][1] = wmma_bf16(ak1, bf1, aK[1][1]);
        aQ[0][0] = wmma_bf16(aq0, bf0, aQ[0][0]);
        aQ[0][1] = wmma_bf16(aq0, bf1, aQ[0][1]);
        aQ[1][0] = wmma_bf16(aq1, bf0, aQ[1][0]);
        aQ[1][1] = wmma_bf16(aq1, bf1, aQ[1][1]);
    }

    int nn = lane & 15, kh = lane >> 4;
    for (int mi = 0; mi < 2; ++mi) {
        #pragma unroll
        for (int r = 0; r < 8; ++r) {
            int row = i0 + 16 * mi + 8 * kh + r;
            float bkv = bk[row], bqv = bq[row];
            for (int nj = 0; nj < 2; ++nj) {
                int col = b0 + 16 * nj + nn;
                Kn[(size_t)row * BD + col] = (__bf16)(aK[mi][nj][r] + bkv);
                Qn[(size_t)row * BD + col] = (__bf16)(aQ[mi][nj][r] + bqv);
            }
        }
    }
}

// ---------------------------------------------------------------- fp32 row sum-of-squares (bf16 [rows][BD])
__global__ void rowsumsq_kernel(const __bf16* __restrict__ M, float* __restrict__ out) {
    int lane = threadIdx.x & 31;
    size_t row = (size_t)blockIdx.x * (blockDim.x >> 5) + (threadIdx.x >> 5);
    const __bf16* p = M + row * BD;
    float s = 0.f;
    for (int b = lane * 4; b < BD; b += 128) {
        v4bf v = *reinterpret_cast<const v4bf*>(p + b);
        float f0 = (float)v[0], f1 = (float)v[1], f2 = (float)v[2], f3 = (float)v[3];
        s += f0 * f0 + f1 * f1 + f2 * f2 + f3 * f3;
    }
    for (int off = 16; off > 0; off >>= 1) s += __shfl_xor(s, off, 32);
    if (lane == 0) out[row] = s;
}

// ---------------------------------------------------------------- TDM double-buffered block GEMM
// Out[row][col] = sum_k A[row][k] * Bt[col][k], all 2048x2048 per batch.
// Block (8 waves) computes 128x128; waves in 2x4 grid each own 64x32.
// NORM: Out *= rsqrt(max(dA[row]*dB[col], eps))  (cosine scores); else plain f32 store.
template <bool NORM>
__global__ void __launch_bounds__(256) gemm_lds_kernel(
    const __bf16* __restrict__ A, const __bf16* __restrict__ Bt,
    const float* __restrict__ dA, const float* __restrict__ dB,
    float* __restrict__ Out) {
    __shared__ __attribute__((aligned(16))) __bf16 smA[2][128 * LDS_STRIDE];
    __shared__ __attribute__((aligned(16))) __bf16 smB[2][128 * LDS_STRIDE];

    int nbat = blockIdx.y;
    const __bf16* An = A + (size_t)nbat * 2048 * 2048;
    const __bf16* Bn = Bt + (size_t)nbat * 2048 * 2048;
    float* On = Out + (size_t)nbat * 2048 * 2048;
    const float* dAn = dA ? dA + nbat * CD : nullptr;
    const float* dBn = dB ? dB + nbat * CD : nullptr;

    int row0 = (blockIdx.x >> 4) * 128;
    int col0 = (blockIdx.x & 15) * 128;
    int lane = threadIdx.x & 31;
    int w = threadIdx.x >> 5;
    int wr = (w >> 2) * 64;   // wave row offset in block tile
    int wc = (w & 3) * 32;    // wave col offset in block tile

    v8f acc[4][2];
    for (int mi = 0; mi < 4; ++mi)
        for (int nj = 0; nj < 2; ++nj) acc[mi][nj] = {};

    // prologue: stage k-step 0 into buffer 0
    if (threadIdx.x < 32) {
        tdm_load_tile(An + (size_t)row0 * 2048, &smA[0][0], 128, 2048);
        tdm_load_tile(Bn + (size_t)col0 * 2048, &smB[0][0], 128, 2048);
    }

    const int KSTEPS = 2048 / 32;
    for (int ks = 0; ks < KSTEPS; ++ks) {
        int cur = ks & 1;
        if (threadIdx.x < 32) {
            if (ks + 1 < KSTEPS) {
                int k0n = (ks + 1) * 32;
                tdm_load_tile(An + (size_t)row0 * 2048 + k0n, &smA[cur ^ 1][0], 128, 2048);
                tdm_load_tile(Bn + (size_t)col0 * 2048 + k0n, &smB[cur ^ 1][0], 128, 2048);
                __builtin_amdgcn_s_wait_tensorcnt(2);   // in-order => current buffer done
            } else {
                __builtin_amdgcn_s_wait_tensorcnt(0);
            }
        }
        __syncthreads();   // current buffer visible to all waves

        const __bf16* pA = &smA[cur][0];
        const __bf16* pB = &smB[cur][0];
        v16bf af[4], bfr[2];
        #pragma unroll
        for (int mi = 0; mi < 4; ++mi) af[mi] = lds_frag_a(pA, wr + 16 * mi, lane);
        #pragma unroll
        for (int nj = 0; nj < 2; ++nj) bfr[nj] = lds_frag_b(pB, wc + 16 * nj, lane);
        #pragma unroll
        for (int mi = 0; mi < 4; ++mi)
            #pragma unroll
            for (int nj = 0; nj < 2; ++nj)
                acc[mi][nj] = wmma_bf16(af[mi], bfr[nj], acc[mi][nj]);

        __syncthreads();   // all reads done before DMA may overwrite this buffer
    }

    int nn = lane & 15, kh = lane >> 4;
    for (int mi = 0; mi < 4; ++mi) {
        #pragma unroll
        for (int r = 0; r < 8; ++r) {
            int row = row0 + wr + 16 * mi + 8 * kh + r;
            float dqv = NORM ? dAn[row] : 0.f;
            for (int nj = 0; nj < 2; ++nj) {
                int col = col0 + wc + 16 * nj + nn;
                float v = acc[mi][nj][r];
                if (NORM) {
                    float d = fmaxf(dqv * dBn[col], 1e-12f);
                    v *= rsqrtf(d);
                }
                On[(size_t)row * 2048 + col] = v;
            }
        }
    }
}

// ---------------------------------------------------------------- online softmax stats per column j
__global__ void colstats_kernel(const float* __restrict__ Y,
                                float* __restrict__ cmax, float* __restrict__ csum) {
    int nbat = blockIdx.y;
    const float* Yn = Y + (size_t)nbat * CD * CD;
    int j = blockIdx.x * blockDim.x + threadIdx.x;
    float m = -1e30f, s = 0.f;
    for (int i = 0; i < CD; ++i) {
        float v = Yn[(size_t)i * CD + j];
        if (v > m) { s *= __expf(m - v); m = v; }
        s += __expf(v - m);
    }
    cmax[nbat * CD + j] = m;
    csum[nbat * CD + j] = s;
}

// ---------------------------------------------------------------- SMt[j][i] = exp(Y[i][j]-max_j)/sum_j (bf16, transposed)
__global__ void norm_transpose_kernel(const float* __restrict__ Y,
                                      const float* __restrict__ cmax,
                                      const float* __restrict__ csum,
                                      __bf16* __restrict__ SMt) {
    __shared__ float tile[32][33];
    int nbat = blockIdx.y;
    const float* Yn = Y + (size_t)nbat * CD * CD;
    __bf16* Sn = SMt + (size_t)nbat * CD * CD;
    const int tiles_x = CD / 32;
    int j0 = (blockIdx.x % tiles_x) * 32;
    int i0 = (blockIdx.x / tiles_x) * 32;
    int jj = j0 + threadIdx.x;
    float cm = cmax[nbat * CD + jj];
    float inv = 1.f / csum[nbat * CD + jj];
    for (int dy = 0; dy < 32; dy += 8) {
        float v = Yn[(size_t)(i0 + threadIdx.y + dy) * CD + jj];
        tile[threadIdx.y + dy][threadIdx.x] = __expf(v - cm) * inv;
    }
    __syncthreads();
    for (int dy = 0; dy < 32; dy += 8)
        Sn[(size_t)(j0 + threadIdx.y + dy) * CD + i0 + threadIdx.x] =
            (__bf16)tile[threadIdx.x][threadIdx.y + dy];
}

// ---------------------------------------------------------------- launch
extern "C" void kernel_launch(void* const* d_in, const int* in_sizes, int n_in,
                              void* d_out, int out_size, void* d_ws, size_t ws_size,
                              hipStream_t stream) {
    const float* X   = (const float*)d_in[0];
    const float* Wk  = (const float*)d_in[1];
    const float* Wq  = (const float*)d_in[2];
    const float* Wk0 = (const float*)d_in[3];
    const float* Wq0 = (const float*)d_in[4];
    float* out = (float*)d_out;

    char* ws = (char*)d_ws;
    size_t off = 0;
    auto take = [&](size_t bytes) -> char* {
        char* p = ws + off;
        off += (bytes + 255) & ~(size_t)255;
        return p;
    };
    __bf16* Xt  = (__bf16*)take((size_t)NB * BD * CD * 2);  //  64 MB
    __bf16* Wkb = (__bf16*)take((size_t)CD * CD * 2);       //   8 MB
    __bf16* Wqb = (__bf16*)take((size_t)CD * CD * 2);       //   8 MB
    __bf16* Kb  = (__bf16*)take((size_t)NB * CD * BD * 2);  //  64 MB
    __bf16* Qb  = (__bf16*)take((size_t)NB * CD * BD * 2);  //  64 MB
    float* DK2  = (float*)take((size_t)NB * CD * 4);
    float* DQ2  = (float*)take((size_t)NB * CD * 4);
    float* cmax = (float*)take((size_t)NB * CD * 4);
    float* csum = (float*)take((size_t)NB * CD * 4);
    __bf16* SMt = Qb;   // alias: Qb dead once scores are computed
    float* Y = out;     // scores alias d_out; dead before Z is written

    // 1) casts + transpose
    cast_bf16_kernel<<<(CD * CD) / 256, 256, 0, stream>>>(Wk, Wkb, CD * CD);
    cast_bf16_kernel<<<(CD * CD) / 256, 256, 0, stream>>>(Wq, Wqb, CD * CD);
    {
        dim3 grid((BD / 32) * (CD / 32), NB), block(32, 8);
        transpose_cast_kernel<<<grid, block, 0, stream>>>(X, Xt);
    }
    // 2) K,Q projection GEMM
    {
        dim3 grid((CD / 32) * (BD / 32) / 8, NB);
        gemm_kq_kernel<<<grid, 256, 0, stream>>>(Wkb, Wqb, Xt, Wk0, Wq0, Kb, Qb);
    }
    // 3) row norms
    rowsumsq_kernel<<<(NB * CD) / 8, 256, 0, stream>>>(Kb, DK2);
    rowsumsq_kernel<<<(NB * CD) / 8, 256, 0, stream>>>(Qb, DQ2);
    // 4) cosine scores (TDM + LDS path)
    {
        dim3 grid((CD / 128) * (CD / 128), NB);
        gemm_lds_kernel<true><<<grid, 256, 0, stream>>>(Qb, Kb, DQ2, DK2, Y);
    }
    // 5) softmax over query-channel axis (per column), emitted transposed
    {
        dim3 grid(CD / 256, NB);
        colstats_kernel<<<grid, 256, 0, stream>>>(Y, cmax, csum);
    }
    {
        dim3 grid((CD / 32) * (CD / 32), NB), block(32, 8);
        norm_transpose_kernel<<<grid, block, 0, stream>>>(Y, cmax, csum, SMt);
    }
    // 6) Z = SM^T * X (TDM + LDS path)
    {
        dim3 grid((CD / 128) * (BD / 128), NB);
        gemm_lds_kernel<false><<<grid, 256, 0, stream>>>(SMt, Xt, nullptr, nullptr, out);
    }
}
